// Attention_2010044694916
// MI455X (gfx1250) — compile-verified, matrix operand order checked by reading, not run on hardware
//
#include <hip/hip_runtime.h>
#include <hip/hip_bf16.h>

// ---------------------------------------------------------------------------
// Problem constants
// ---------------------------------------------------------------------------
#define BATCH 128
#define NH    12
#define DIM   768
#define NTOK  245          // 7*7 + 14*14
#define NPAD  256          // padded token count (multiple of 32)
#define DH    64           // head dim
#define MROWS (BATCH * NTOK)   // 31360 rows for the two big GEMMs
#define VTLD  40           // padded row stride of V^T tile (16B aligned, bank-spread)

typedef __attribute__((ext_vector_type(16))) __bf16 v16bf;
typedef __attribute__((ext_vector_type(8)))  __bf16 v8bf;
typedef __attribute__((ext_vector_type(4)))  __bf16 v4bf;
typedef __attribute__((ext_vector_type(8)))  float  v8f;

// CDNA5 WMMA bf16: D = A(16x32) * B(32x16) + C(16x16 f32)
#define WMMA_BF16(A, B, C) \
  __builtin_amdgcn_wmma_f32_16x16x32_bf16(false, (A), false, (B), (short)0, (C), false, false)

// ---------------------------------------------------------------------------
// CDNA5 async global->LDS copies (ASYNCcnt-tracked).  LDS address = low 32
// bits of the generic pointer (the AS3 offset); global address = 64-bit VGPR
// pair.  Paired with s_wait_asynccnt before the workgroup barrier.
// ---------------------------------------------------------------------------
__device__ __forceinline__ void async_copy_b128(void* lds, const void* g) {
  unsigned lo = (unsigned)(size_t)lds;
  unsigned long long ga = (unsigned long long)(size_t)g;
  asm volatile("global_load_async_to_lds_b128 %0, %1, off"
               :: "v"(lo), "v"(ga) : "memory");
}
__device__ __forceinline__ void async_copy_b64(void* lds, const void* g) {
  unsigned lo = (unsigned)(size_t)lds;
  unsigned long long ga = (unsigned long long)(size_t)g;
  asm volatile("global_load_async_to_lds_b64 %0, %1, off"
               :: "v"(lo), "v"(ga) : "memory");
}
__device__ __forceinline__ void wait_async() {
  asm volatile("s_wait_asynccnt 0x0" ::: "memory");
}

// Fragment loader for 16-bit A/B operands.
// Per ISA 7.12.2: lanes 0-15 hold K = {0..7, 16..23}, lanes 16-31 hold
// K = {8..15, 24..31} of their row/col.  Two 16-byte contiguous loads.
__device__ __forceinline__ v16bf load_frag16(const __bf16* __restrict__ base,
                                             int row, int ld, int half) {
  const __bf16* p0 = base + row * ld + half * 8;
  const __bf16* p1 = base + row * ld + 16 + half * 8;
  v8bf lo = *(const v8bf*)p0;
  v8bf hi = *(const v8bf*)p1;
  v16bf r;
#pragma unroll
  for (int i = 0; i < 8; ++i) { r[i] = lo[i]; r[i + 8] = hi[i]; }
  return r;
}

// ---------------------------------------------------------------------------
// K0: fp32 -> bf16 cast (weights only; x is converted in-GEMM)
// ---------------------------------------------------------------------------
__global__ void cast_f32_bf16(const float* __restrict__ src,
                              __bf16* __restrict__ dst, int n) {
  int i = blockIdx.x * blockDim.x + threadIdx.x;
  int stride = gridDim.x * blockDim.x;
  for (; i < n; i += stride) dst[i] = (__bf16)src[i];
}

// ---------------------------------------------------------------------------
// K1: zero padded rows (tok 245..255) of q/k/v so WMMA never eats stale bytes
// ---------------------------------------------------------------------------
__global__ void zero_pad_rows(__bf16* __restrict__ q, __bf16* __restrict__ k,
                              __bf16* __restrict__ v) {
  const int tot = BATCH * NH * (NPAD - NTOK) * DH;
  int i = blockIdx.x * blockDim.x + threadIdx.x;
  if (i >= tot) return;
  int d  = i % DH;
  int r  = (i / DH) % (NPAD - NTOK);
  int bh = i / (DH * (NPAD - NTOK));
  size_t off = ((size_t)bh * NPAD + NTOK + r) * DH + d;
  q[off] = (__bf16)0.f; k[off] = (__bf16)0.f; v[off] = (__bf16)0.f;
}

// ---------------------------------------------------------------------------
// K2: expand bias_table[h, rel_index] into padded (H, 256, 256) fp32 matrix.
// Padded key columns get -1e30 => doubles as the softmax mask.
// ---------------------------------------------------------------------------
__global__ void build_bias(const float* __restrict__ bt,
                           const int* __restrict__ ri,
                           float* __restrict__ bm, int n_bias) {
  int i = blockIdx.x * blockDim.x + threadIdx.x;      // H*256*256 total
  int h  = i >> 16;
  int qk = i & 0xFFFF;
  int qr = qk >> 8;
  int kc = qk & 0xFF;
  float val = -1e30f;
  if (qr < NTOK && kc < NTOK) val = bt[h * n_bias + ri[qr * NTOK + kc]];
  bm[i] = val;
}

// ---------------------------------------------------------------------------
// K3: QKV projection GEMM  (MROWS x 768 fp32) @ (2304 x 768)^T bf16, WMMA.
// x stays fp32 in HBM/L2 and is converted to bf16 while staging into LDS
// (saves a full 150 MB cast pass).  Weight tile staged with async copies.
// Tile: 128 (M) x 64 (N) per 256-thread block; 8 waves, 2x2 16x16 tiles each.
// Epilogue scatters into padded (B,H,256,64) q/k/v (q pre-scaled by Dh^-1/2).
// ---------------------------------------------------------------------------
__global__ __launch_bounds__(256)
void qkv_gemm(const float* __restrict__ X, const __bf16* __restrict__ W,
              __bf16* __restrict__ Q, __bf16* __restrict__ K,
              __bf16* __restrict__ V) {
  __shared__ __align__(32) __bf16 As[128 * 32];   // [m][k]
  __shared__ __align__(32) __bf16 Bs[64 * 32];    // [n][k]  (W row-major [o][c])

  const int t    = threadIdx.x;
  const int lane = t & 31;
  const int wid  = t >> 5;            // 0..7
  const int wm   = wid >> 1;          // 0..3  -> 32-row band
  const int wn   = wid & 1;           // 0..1  -> 32-col band
  const int half = lane >> 4, l16 = lane & 15;
  const int m0 = blockIdx.x * 128;
  const int n0 = blockIdx.y * 64;

  const int a_row = t >> 1, a_off = (t & 1) * 16;  // 16 elems per thread
  const int b_row = t >> 2, b_off = (t & 3) * 8;   // 8 elems (16B) per thread

  v8f acc[2][2] = {};

  for (int k0 = 0; k0 < DIM; k0 += 32) {
    __syncthreads();
    // A tile: fp32 global load -> bf16 convert -> LDS
    {
      const float* ga = X + (size_t)(m0 + a_row) * DIM + k0 + a_off;
      v8f f0 = *(const v8f*)ga;
      v8f f1 = *(const v8f*)(ga + 8);
      v16bf ab;
#pragma unroll
      for (int i = 0; i < 8; ++i) {
        ab[i] = (__bf16)f0[i];
        ab[i + 8] = (__bf16)f1[i];
      }
      *(v16bf*)&As[a_row * 32 + a_off] = ab;
    }
    // B tile: async global->LDS (already bf16)
    async_copy_b128(&Bs[b_row * 32 + b_off],
                    W + (size_t)(n0 + b_row) * DIM + k0 + b_off);
    if (k0 + 32 < DIM) {  // prefetch next A slice -> global_prefetch_b8
      __builtin_prefetch(X + (size_t)(m0 + a_row) * DIM + k0 + 32 + a_off, 0, 0);
    }
    wait_async();
    __syncthreads();

    v16bf af[2], bf[2];
#pragma unroll
    for (int rt = 0; rt < 2; ++rt)
      af[rt] = load_frag16(&As[(wm * 32 + rt * 16) * 32], l16, 32, half);
#pragma unroll
    for (int ct = 0; ct < 2; ++ct)
      bf[ct] = load_frag16(&Bs[(wn * 32 + ct * 16) * 32], l16, 32, half);
#pragma unroll
    for (int rt = 0; rt < 2; ++rt)
#pragma unroll
      for (int ct = 0; ct < 2; ++ct)
        acc[rt][ct] = WMMA_BF16(af[rt], bf[ct], acc[rt][ct]);
  }

  // Epilogue: col O in [0,2304): s = O/768 picks q/k/v, h = (O%768)/64, d = O%64.
#pragma unroll
  for (int rt = 0; rt < 2; ++rt) {
#pragma unroll
    for (int ct = 0; ct < 2; ++ct) {
      int colO = n0 + wn * 32 + ct * 16 + l16;
      int s    = colO / DIM;
      int rem  = colO - s * DIM;
      int h    = rem >> 6;
      int d    = rem & 63;
      __bf16* dst = (s == 0) ? Q : ((s == 1) ? K : V);
      float scale = (s == 0) ? 0.125f : 1.0f;   // Dh^-0.5 folded into q
#pragma unroll
      for (int r = 0; r < 8; ++r) {
        int rowM = m0 + wm * 32 + rt * 16 + half * 8 + r;   // always < MROWS
        int b    = rowM / NTOK;
        int tok  = rowM - b * NTOK;
        dst[((size_t)(b * NH + h) * NPAD + tok) * DH + d] =
            (__bf16)(acc[rt][ct][r] * scale);
      }
    }
  }
}

// ---------------------------------------------------------------------------
// K4: fused flash attention. One block per (b,h); 512 threads = 16 waves;
// wave w owns query rows [16w, 16w+16). 8 key tiles of 32 keys.
// K tile async-copied (contiguous 4KB in padded layout); V tile transposed
// through registers (coalesced b64 reads, b16 LDS scatter into padded V^T).
// ---------------------------------------------------------------------------
__global__ __launch_bounds__(512)
void attn_kernel(const __bf16* __restrict__ Q, const __bf16* __restrict__ K,
                 const __bf16* __restrict__ V, const float* __restrict__ BM,
                 __bf16* __restrict__ O) {
  __shared__ __align__(32) __bf16 Ks[32 * 64];       // [key][d], contiguous
  __shared__ __align__(32) __bf16 VTs[64 * VTLD];    // [d][key], padded rows
  __shared__ __align__(32) __bf16 Ps[16][16 * 32];   // per-wave P staging

  const int bh = blockIdx.x;
  const int h  = bh % NH;
  const int b  = bh / NH;
  const int t    = threadIdx.x;
  const int wid  = t >> 5;                // query tile 0..15
  const int lane = t & 31;
  const int half = lane >> 4, l16 = lane & 15;
  const int qt0  = wid * 16;

  const __bf16* Qb = Q + (size_t)bh * NPAD * DH;
  const __bf16* Kb = K + (size_t)bh * NPAD * DH;
  const __bf16* Vb = V + (size_t)bh * NPAD * DH;
  const float*  bm = BM + (size_t)h * NPAD * NPAD;

  // Q fragments for this wave's 16 rows, two 32-wide d-windows.
  v16bf qf[2];
#pragma unroll
  for (int dw = 0; dw < 2; ++dw)
    qf[dw] = load_frag16(Qb + (size_t)qt0 * DH + dw * 32, l16, DH, half);

  v8f acc[4] = {};
  float mrow[8], lrow[8];
#pragma unroll
  for (int r = 0; r < 8; ++r) { mrow[r] = -1e30f; lrow[r] = 0.f; }

  const int idx = t * 4;                  // 4 bf16 per thread per tile
  const int kr = idx >> 6, kc = idx & 63;

  for (int kt = 0; kt < NPAD; kt += 32) {
    __syncthreads();                      // previous tile fully consumed
    // K tile: rows contiguous in padded layout -> flat 4KB async copy
    async_copy_b64(&Ks[idx], Kb + (size_t)kt * DH + idx);
    // V^T tile: coalesced global read, transposed LDS scatter
    {
      v4bf vv = *(const v4bf*)(Vb + (size_t)kt * DH + idx);
#pragma unroll
      for (int j = 0; j < 4; ++j) VTs[(kc + j) * VTLD + kr] = vv[j];
    }
    wait_async();
    __syncthreads();

    // ---- scores: C initialized with (bias + mask) => S = QK^T*scale + bias
    v8f s0, s1;
#pragma unroll
    for (int r = 0; r < 8; ++r) {
      int row = qt0 + half * 8 + r;
      s0[r] = bm[row * NPAD + kt + l16];
      s1[r] = bm[row * NPAD + kt + 16 + l16];
    }
#pragma unroll
    for (int dw = 0; dw < 2; ++dw) {
      v16bf kf0 = load_frag16(&Ks[dw * 32], l16, 64, half);            // keys kt..+15
      v16bf kf1 = load_frag16(&Ks[16 * 64 + dw * 32], l16, 64, half);  // keys +16..+31
      s0 = WMMA_BF16(qf[dw], kf0, s0);
      s1 = WMMA_BF16(qf[dw], kf1, s1);
    }

    // ---- online softmax (rows live across 16 lanes of a half-wave)
    float p0[8], p1[8], fac[8];
#pragma unroll
    for (int r = 0; r < 8; ++r) {
      float mx = fmaxf(s0[r], s1[r]);
#pragma unroll
      for (int off = 1; off < 16; off <<= 1)
        mx = fmaxf(mx, __shfl_xor(mx, off, 32));
      float mnew = fmaxf(mrow[r], mx);
      fac[r] = __expf(mrow[r] - mnew);
      mrow[r] = mnew;
      p0[r] = __expf(s0[r] - mnew);
      p1[r] = __expf(s1[r] - mnew);
      float rs = p0[r] + p1[r];
#pragma unroll
      for (int off = 1; off < 16; off <<= 1)
        rs += __shfl_xor(rs, off, 32);
      lrow[r] = lrow[r] * fac[r] + rs;
    }
#pragma unroll
    for (int dt = 0; dt < 4; ++dt)
#pragma unroll
      for (int r = 0; r < 8; ++r) acc[dt][r] *= fac[r];

    // ---- P: C-layout -> per-wave LDS -> A-fragment
#pragma unroll
    for (int r = 0; r < 8; ++r) {
      int row = half * 8 + r;
      Ps[wid][row * 32 + l16]      = (__bf16)p0[r];
      Ps[wid][row * 32 + 16 + l16] = (__bf16)p1[r];
    }
    v16bf pf = load_frag16(&Ps[wid][0], l16, 32, half);

    // ---- O += P(16x32) @ V(32x64), one WMMA per 16-wide d tile
#pragma unroll
    for (int dt = 0; dt < 4; ++dt) {
      v16bf vf = load_frag16(&VTs[0], dt * 16 + l16, VTLD, half);
      acc[dt] = WMMA_BF16(pf, vf, acc[dt]);
    }
  }

  // ---- finalize: O / l, write bf16 to (B*N, 768) at col h*64+d
#pragma unroll
  for (int dt = 0; dt < 4; ++dt) {
#pragma unroll
    for (int r = 0; r < 8; ++r) {
      int row = qt0 + half * 8 + r;
      if (row < NTOK) {
        O[(size_t)(b * NTOK + row) * DIM + h * DH + dt * 16 + l16] =
            (__bf16)(acc[dt][r] / lrow[r]);
      }
    }
  }
}

// ---------------------------------------------------------------------------
// K5: output projection (MROWS x 768) @ (768 x 768)^T + bias -> fp32 d_out.
// Both tiles are bf16 => both staged with async copies.
// ---------------------------------------------------------------------------
__global__ __launch_bounds__(256)
void proj_gemm(const __bf16* __restrict__ X, const __bf16* __restrict__ W,
               const float* __restrict__ bias, float* __restrict__ out) {
  __shared__ __align__(32) __bf16 As[128 * 32];
  __shared__ __align__(32) __bf16 Bs[64 * 32];

  const int t    = threadIdx.x;
  const int lane = t & 31;
  const int wid  = t >> 5;
  const int wm   = wid >> 1;
  const int wn   = wid & 1;
  const int half = lane >> 4, l16 = lane & 15;
  const int m0 = blockIdx.x * 128;
  const int n0 = blockIdx.y * 64;

  const int a_row = t >> 1, a_off = (t & 1) * 16;
  const int b_row = t >> 2, b_off = (t & 3) * 8;

  v8f acc[2][2] = {};

  for (int k0 = 0; k0 < DIM; k0 += 32) {
    __syncthreads();
    // A tile: two async b128 per thread (32B)
    async_copy_b128(&As[a_row * 32 + a_off],
                    X + (size_t)(m0 + a_row) * DIM + k0 + a_off);
    async_copy_b128(&As[a_row * 32 + a_off + 8],
                    X + (size_t)(m0 + a_row) * DIM + k0 + a_off + 8);
    // B tile: one async b128 per thread
    async_copy_b128(&Bs[b_row * 32 + b_off],
                    W + (size_t)(n0 + b_row) * DIM + k0 + b_off);
    if (k0 + 32 < DIM)
      __builtin_prefetch(X + (size_t)(m0 + a_row) * DIM + k0 + 32 + a_off, 0, 0);
    wait_async();
    __syncthreads();

    v16bf af[2], bf[2];
#pragma unroll
    for (int rt = 0; rt < 2; ++rt)
      af[rt] = load_frag16(&As[(wm * 32 + rt * 16) * 32], l16, 32, half);
#pragma unroll
    for (int ct = 0; ct < 2; ++ct)
      bf[ct] = load_frag16(&Bs[(wn * 32 + ct * 16) * 32], l16, 32, half);
#pragma unroll
    for (int rt = 0; rt < 2; ++rt)
#pragma unroll
      for (int ct = 0; ct < 2; ++ct)
        acc[rt][ct] = WMMA_BF16(af[rt], bf[ct], acc[rt][ct]);
  }

#pragma unroll
  for (int rt = 0; rt < 2; ++rt) {
#pragma unroll
    for (int ct = 0; ct < 2; ++ct) {
      int colO = n0 + wn * 32 + ct * 16 + l16;
      float bv = bias[colO];
#pragma unroll
      for (int r = 0; r < 8; ++r) {
        int rowM = m0 + wm * 32 + rt * 16 + half * 8 + r;
        out[(size_t)rowM * DIM + colO] = acc[rt][ct][r] + bv;
      }
    }
  }
}

// ---------------------------------------------------------------------------
// Host-side launcher
// ---------------------------------------------------------------------------
extern "C" void kernel_launch(void* const* d_in, const int* in_sizes, int n_in,
                              void* d_out, int out_size, void* d_ws, size_t ws_size,
                              hipStream_t stream) {
  const float* x          = (const float*)d_in[0];
  const float* qkv_w      = (const float*)d_in[1];
  const float* proj_w     = (const float*)d_in[2];
  const float* proj_b     = (const float*)d_in[3];
  const float* bias_table = (const float*)d_in[4];
  const int*   rel_index  = (const int*)d_in[5];
  const int n_bias = in_sizes[4] / NH;

  char* ws = (char*)d_ws;
  size_t off = 0;
  auto carve = [&](size_t bytes) -> void* {
    void* p = ws + off;
    off += (bytes + 255) & ~(size_t)255;
    return p;
  };
  __bf16* wq_bf = (__bf16*)carve((size_t)3 * DIM * DIM * 2);
  __bf16* wp_bf = (__bf16*)carve((size_t)DIM * DIM * 2);
  __bf16* q_bf  = (__bf16*)carve((size_t)BATCH * NH * NPAD * DH * 2);
  __bf16* k_bf  = (__bf16*)carve((size_t)BATCH * NH * NPAD * DH * 2);
  __bf16* v_bf  = (__bf16*)carve((size_t)BATCH * NH * NPAD * DH * 2);
  __bf16* ao_bf = (__bf16*)carve((size_t)MROWS * DIM * 2);
  float*  biasm = (float*)carve((size_t)NH * NPAD * NPAD * 4);

  // weight casts (small)
  cast_f32_bf16<<<1024, 256, 0, stream>>>(qkv_w, wq_bf, 3 * DIM * DIM);
  cast_f32_bf16<<<512, 256, 0, stream>>>(proj_w, wp_bf, DIM * DIM);

  // pad rows + bias matrix
  {
    int tot = BATCH * NH * (NPAD - NTOK) * DH;
    zero_pad_rows<<<(tot + 255) / 256, 256, 0, stream>>>(q_bf, k_bf, v_bf);
  }
  build_bias<<<(NH * NPAD * NPAD) / 256, 256, 0, stream>>>(bias_table, rel_index,
                                                           biasm, n_bias);

  // qkv projection: M tiles = 31360/128 = 245, N tiles = 2304/64 = 36
  qkv_gemm<<<dim3(MROWS / 128, (3 * DIM) / 64), 256, 0, stream>>>(
      x, wq_bf, q_bf, k_bf, v_bf);

  // fused attention: one block per (b, h)
  attn_kernel<<<BATCH * NH, 512, 0, stream>>>(q_bf, k_bf, v_bf, biasm, ao_bf);

  // output projection: N tiles = 768/64 = 12
  proj_gemm<<<dim3(MROWS / 128, DIM / 64), 256, 0, stream>>>(
      ao_bf, wp_bf, proj_b, (float*)d_out);
}
